// EncoderLayer_55027120996712
// MI455X (gfx1250) — matrix operand order repaired and checked by reading, not compiled, not run on hardware
//
#include <hip/hip_runtime.h>
#include <hip/hip_bf16.h>
#include <math.h>

// ---------------------------------------------------------------------------
// CDNA5 WMMA types / helpers
// ---------------------------------------------------------------------------
typedef __attribute__((ext_vector_type(16))) _Float16 v16h;
typedef __attribute__((ext_vector_type(8)))  float    v8f;

#define WMMA_F16(a, b, c) \
  __builtin_amdgcn_wmma_f32_16x16x32_f16(false, (a), false, (b), (short)0, (c), false, false)

// Async global->LDS copy (GLOBAL_LOAD_ASYNC_TO_LDS_B128, tracked by ASYNCcnt).
// vdst = per-lane LDS byte offset (low 32 bits of generic LDS pointer,
// per the CDNA5 LDS-aperture rule), vaddr = 64-bit global address.
__device__ __forceinline__ void async_copy_b128(void* lds, const void* gptr) {
  unsigned lds_off = (unsigned)(unsigned long long)lds;
  unsigned long long ga = (unsigned long long)gptr;
  asm volatile("global_load_async_to_lds_b128 %0, %1, off"
               :: "v"(lds_off), "v"(ga) : "memory");
}
__device__ __forceinline__ void wait_async0() {
  asm volatile("s_wait_asynccnt 0x0" ::: "memory");
}

// A-matrix fragment (16x32 f16) from LDS, row-major with given stride (halves).
// Lane L<16 holds row L, K in {0..7, 16..23}; lane L>=16 holds row L-16,
// K in {8..15, 24..31}  (CDNA5 ISA 7.12.2).
__device__ __forceinline__ v16h frag_a_lds(const _Float16* base, int stride,
                                           int row0, int kbase, int lane) {
  const int lr  = lane & 15;
  const int khi = (lane < 16) ? 0 : 8;
  const _Float16* r = base + (size_t)(row0 + lr) * stride + kbase + khi;
  v16h f;
#pragma unroll
  for (int h = 0; h < 8; ++h) f[h] = r[h];
#pragma unroll
  for (int h = 0; h < 8; ++h) f[8 + h] = r[16 + h];
  return f;
}

// B-matrix fragment (32x16 f16): LDS tile stored column-contiguous
// (tile[n][k]), lane n<16 holds column n K in {0..7, 16..23}, lane n+16 holds K in {8..15, 24..31}.
__device__ __forceinline__ v16h frag_b_lds(const _Float16* base, int stride,
                                           int col0, int kbase, int lane) {
  const int ln = lane & 15;
  const int kb = kbase + ((lane < 16) ? 0 : 8);
  const _Float16* r = base + (size_t)(col0 + ln) * stride + kb;
  v16h f;
#pragma unroll
  for (int h = 0; h < 8; ++h) f[h] = r[h];
#pragma unroll
  for (int h = 0; h < 8; ++h) f[8 + h] = r[16 + h];
  return f;
}

// A fragment built from an f32 score tile in LDS ([64][512]), cvt to f16.
__device__ __forceinline__ v16h frag_a_scores(const float* sc, int row0,
                                              int kbase, int lane) {
  const int lr  = lane & 15;
  const int khi = (lane < 16) ? 0 : 8;
  const float* r = sc + (size_t)(row0 + lr) * 512 + kbase + khi;
  v16h f;
#pragma unroll
  for (int h = 0; h < 8; ++h) f[h] = (_Float16)r[h];
#pragma unroll
  for (int h = 0; h < 8; ++h) f[8 + h] = (_Float16)r[16 + h];
  return f;
}

// ---------------------------------------------------------------------------
// fp32 -> f16 conversion, float4 in / 8-byte packed out
// ---------------------------------------------------------------------------
__global__ __launch_bounds__(256) void cvt_f32_to_f16(
    const float* __restrict__ in, _Float16* __restrict__ out, size_t n4) {
  size_t i = (size_t)blockIdx.x * blockDim.x + threadIdx.x;
  size_t stride = (size_t)gridDim.x * blockDim.x;
  for (; i < n4; i += stride) {
    float4 v = ((const float4*)in)[i];
    _Float16 h[4] = {(_Float16)v.x, (_Float16)v.y, (_Float16)v.z, (_Float16)v.w};
    ((uint2*)out)[i] = *(const uint2*)h;
  }
}

// ---------------------------------------------------------------------------
// LDS-tiled WMMA GEMM: C[M,N] = act(A[M,K] @ B[K,N] + bias)
// 128x64 block tile, 8 waves; each wave owns a 32x32 region = 2x2 WMMA tiles.
// A tile filled with async global->LDS b128 copies; B tile via VGPR transpose.
// ---------------------------------------------------------------------------
template <bool WF32, bool WF16, bool GELU>
__global__ __launch_bounds__(256) void gemm_f16_wmma(
    const _Float16* __restrict__ A, const _Float16* __restrict__ B,
    const float* __restrict__ bias, float* __restrict__ outF,
    _Float16* __restrict__ outH, int M, int N, int K) {
  __shared__ _Float16 As[128][40];  // [m][k]
  __shared__ _Float16 Bs[64][40];   // [n][k]  (transposed tile)
  const int tid  = threadIdx.x;
  const int lane = tid & 31, wave = tid >> 5;
  const int m0 = blockIdx.y * 128, n0 = blockIdx.x * 64;
  const int wm = (wave & 3) * 32;   // wave rows: wm, wm+16
  const int wn = (wave >> 2) * 32;  // wave cols: wn, wn+16
  v8f acc[2][2] = {};

  const int lm  = tid >> 2;         // A load: row (iter 2 adds 64)
  const int lkc = (tid & 3) << 3;   //         k offset (8 halves = b128)
  const int lbk = tid >> 3;         // B load: k row 0..31
  const int lbn = (tid & 7) << 3;   //         n offset (8 halves)

  for (int k0 = 0; k0 < K; k0 += 32) {
#pragma unroll
    for (int i = 0; i < 2; ++i) {   // A tile 128x32: async global->LDS b128
      int m = lm + i * 64;
      async_copy_b128(&As[m][lkc], A + (size_t)(m0 + m) * K + k0 + lkc);
    }
    {                               // B tile 32x64: b128 read, transposed fill
      uint4 gv = *(const uint4*)(B + (size_t)(k0 + lbk) * N + n0 + lbn);
      const _Float16* hv = (const _Float16*)&gv;
#pragma unroll
      for (int j = 0; j < 8; ++j) Bs[lbn + j][lbk] = hv[j];
    }
    if (k0 + 32 < K) {              // next-tile prefetch -> global_prefetch_b8
      __builtin_prefetch(A + (size_t)(m0 + lm) * K + k0 + 32 + lkc, 0, 0);
      __builtin_prefetch(B + (size_t)(k0 + 32 + lbk) * N + n0 + lbn, 0, 0);
    }
    wait_async0();
    __syncthreads();
    v16h a0 = frag_a_lds(&As[0][0], 40, wm, 0, lane);
    v16h a1 = frag_a_lds(&As[0][0], 40, wm + 16, 0, lane);
    v16h b0 = frag_b_lds(&Bs[0][0], 40, wn, 0, lane);
    v16h b1 = frag_b_lds(&Bs[0][0], 40, wn + 16, 0, lane);
    acc[0][0] = WMMA_F16(a0, b0, acc[0][0]);
    acc[0][1] = WMMA_F16(a0, b1, acc[0][1]);
    acc[1][0] = WMMA_F16(a1, b0, acc[1][0]);
    acc[1][1] = WMMA_F16(a1, b1, acc[1][1]);
    __syncthreads();
  }

  const int moff = (lane < 16) ? 0 : 8;
  const int nl   = lane & 15;
#pragma unroll
  for (int tj = 0; tj < 2; ++tj) {
    const int n  = n0 + wn + tj * 16 + nl;
    const float bv = bias[n];
#pragma unroll
    for (int ti = 0; ti < 2; ++ti) {
      const int mbase = m0 + wm + ti * 16 + moff;
#pragma unroll
      for (int r = 0; r < 8; ++r) {
        float v = acc[ti][tj][r] + bv;
        if (GELU) v = 0.5f * v * (1.0f + erff(v * 0.70710678118654752f));
        size_t idx = (size_t)(mbase + r) * N + n;
        if (WF32) outF[idx] = v;
        if (WF16) outH[idx] = (_Float16)v;
      }
    }
  }
}

// ---------------------------------------------------------------------------
// Attention: one block per (b, h, 64-row q tile).
// scores(64x512 f32 in LDS) = q k^T / 8 + adj + mask*(-1e9); softmax; ctx.
// q/k tiles via async global->LDS copies; v tile via VGPR transpose.
// ---------------------------------------------------------------------------
__global__ __launch_bounds__(256) void attention_wmma(
    const _Float16* __restrict__ qh, const _Float16* __restrict__ kh,
    const _Float16* __restrict__ vh, const float* __restrict__ adj,
    const int* __restrict__ mask, _Float16* __restrict__ ctxh) {
  constexpr int S = 512, D = 1024, DK = 64;
  __shared__ float    sc[64][512];
  __shared__ _Float16 qt[64][72];
  __shared__ _Float16 kv[64][72];
  __shared__ float    red[64][4];
  const int tid  = threadIdx.x;
  const int lane = tid & 31, wave = tid >> 5;
  const int q0 = blockIdx.x * 64;
  const int h  = blockIdx.y;
  const int b  = blockIdx.z;
  const size_t rowbase = (size_t)b * S;
  const int hoff = h * DK;
  const int tm = (wave & 3) * 16;
  const int tn = (wave >> 2) * 32;
  const int moff = (lane < 16) ? 0 : 8;
  const int nl   = lane & 15;

  // q tile (64x64): async global->LDS b128 copies
#pragma unroll
  for (int i = 0; i < 2; ++i) {
    int u = tid + i * 256;
    int m = u >> 3, dc = (u & 7) << 3;
    async_copy_b128(&qt[m][dc], qh + (rowbase + q0 + m) * D + hoff + dc);
  }

  // ---- scores: loop over 8 key chunks of 64 ----
  for (int j = 0; j < 8; ++j) {
#pragma unroll
    for (int i = 0; i < 2; ++i) {           // kv[key][d]: async copies
      int u = tid + i * 256;
      int key = u >> 3, dc = (u & 7) << 3;
      async_copy_b128(&kv[key][dc],
                      kh + (rowbase + j * 64 + key) * D + hoff + dc);
    }
    if (j < 7) {
      int key = tid >> 3, dc = (tid & 7) << 3;
      __builtin_prefetch(kh + (rowbase + (j + 1) * 64 + key) * D + hoff + dc,
                         0, 0);
    }
    wait_async0();
    __syncthreads();
    v8f s0 = {}, s1 = {};
#pragma unroll
    for (int ks = 0; ks < 2; ++ks) {        // K = dk = 64 -> 2 wmma steps
      v16h a  = frag_a_lds(&qt[0][0], 72, tm, ks * 32, lane);
      v16h b0 = frag_b_lds(&kv[0][0], 72, tn, ks * 32, lane);
      v16h b1 = frag_b_lds(&kv[0][0], 72, tn + 16, ks * 32, lane);
      s0 = WMMA_F16(a, b0, s0);
      s1 = WMMA_F16(a, b1, s1);
    }
#pragma unroll
    for (int t = 0; t < 2; ++t) {           // fused scale + adj + mask
      v8f acc = t ? s1 : s0;
      int nloc = tn + t * 16 + nl;
      int key  = j * 64 + nloc;
      float mval = -1e9f * (float)mask[b * S + key];
#pragma unroll
      for (int r = 0; r < 8; ++r) {
        int mloc = tm + moff + r;
        sc[mloc][key] = acc[r] * 0.125f +
                        adj[((size_t)b * S + (q0 + mloc)) * S + key] + mval;
      }
    }
    __syncthreads();
  }

  // ---- softmax over 512 keys per row (4 threads / row, float4 LDS ops) ----
  {
    int row = tid >> 2, part = tid & 3;
    float4* pr = (float4*)&sc[row][part * 128];   // 32 x float4
    float mx = -3.4e38f;
    for (int c = 0; c < 32; ++c) {
      float4 t = pr[c];
      mx = fmaxf(mx, fmaxf(fmaxf(t.x, t.y), fmaxf(t.z, t.w)));
    }
    red[row][part] = mx;
    __syncthreads();
    float rmx = fmaxf(fmaxf(red[row][0], red[row][1]),
                      fmaxf(red[row][2], red[row][3]));
    __syncthreads();
    float sum = 0.0f;
    for (int c = 0; c < 32; ++c) {
      float4 t = pr[c];
      t.x = __expf(t.x - rmx); t.y = __expf(t.y - rmx);
      t.z = __expf(t.z - rmx); t.w = __expf(t.w - rmx);
      pr[c] = t;
      sum += (t.x + t.y) + (t.z + t.w);
    }
    red[row][part] = sum;
    __syncthreads();
    float inv = 1.0f / (red[row][0] + red[row][1] + red[row][2] + red[row][3]);
    for (int c = 0; c < 32; ++c) {
      float4 t = pr[c];
      t.x *= inv; t.y *= inv; t.z *= inv; t.w *= inv;
      pr[c] = t;
    }
    __syncthreads();
  }

  // ---- ctx = attn(64x512) @ v(512x64) ----
  v8f c0 = {}, c1 = {};
  for (int j = 0; j < 8; ++j) {
#pragma unroll
    for (int i = 0; i < 2; ++i) {           // kv[d][key] (transposed v)
      int u = tid + i * 256;
      int key = u >> 3, dc = (u & 7) << 3;
      uint4 gv =
          *(const uint4*)(vh + (rowbase + j * 64 + key) * D + hoff + dc);
      const _Float16* hv = (const _Float16*)&gv;
#pragma unroll
      for (int jj = 0; jj < 8; ++jj) kv[dc + jj][key] = hv[jj];
    }
    if (j < 7) {
      int key = tid >> 3, dc = (tid & 7) << 3;
      __builtin_prefetch(vh + (rowbase + (j + 1) * 64 + key) * D + hoff + dc,
                         0, 0);
    }
    __syncthreads();
#pragma unroll
    for (int ks = 0; ks < 2; ++ks) {
      int kb = j * 64 + ks * 32;
      v16h a  = frag_a_scores(&sc[0][0], tm, kb, lane);
      v16h b0 = frag_b_lds(&kv[0][0], 72, tn, ks * 32, lane);
      v16h b1 = frag_b_lds(&kv[0][0], 72, tn + 16, ks * 32, lane);
      c0 = WMMA_F16(a, b0, c0);
      c1 = WMMA_F16(a, b1, c1);
    }
    __syncthreads();
  }
#pragma unroll
  for (int t = 0; t < 2; ++t) {
    v8f acc = t ? c1 : c0;
    int dloc = tn + t * 16 + nl;
#pragma unroll
    for (int r = 0; r < 8; ++r) {
      int mloc = tm + moff + r;
      ctxh[(rowbase + q0 + mloc) * D + hoff + dloc] = (_Float16)acc[r];
    }
  }
}

// ---------------------------------------------------------------------------
// out = LayerNorm(a + b) with unbiased std (ddof=1) and /(std + eps).
// One block per row (D=1024); exactly one float4 per thread.
// ---------------------------------------------------------------------------
template <bool WF16>
__global__ __launch_bounds__(256) void add_layernorm(
    const float* __restrict__ a, const float* __restrict__ b,
    const float* __restrict__ gamma, const float* __restrict__ beta,
    float* __restrict__ outF, _Float16* __restrict__ outH) {
  constexpr int D = 1024;
  __shared__ float rs[256], rq[256];
  const int tid = threadIdx.x;
  const size_t row = blockIdx.x;
  float4 va = ((const float4*)(a + row * D))[tid];
  float4 vb = ((const float4*)(b + row * D))[tid];
  float v[4] = {va.x + vb.x, va.y + vb.y, va.z + vb.z, va.w + vb.w};
  float s = v[0] + v[1] + v[2] + v[3];
  float q = v[0] * v[0] + v[1] * v[1] + v[2] * v[2] + v[3] * v[3];
  rs[tid] = s; rq[tid] = q;
  __syncthreads();
  for (int off = 128; off > 0; off >>= 1) {
    if (tid < off) { rs[tid] += rs[tid + off]; rq[tid] += rq[tid + off]; }
    __syncthreads();
  }
  float mean = rs[0] * (1.0f / D);
  float var  = (rq[0] - (float)D * mean * mean) * (1.0f / (D - 1)); // ddof=1
  float inv  = 1.0f / (sqrtf(fmaxf(var, 0.0f)) + 1e-6f);
  float4 g4 = ((const float4*)gamma)[tid];
  float4 t4 = ((const float4*)beta)[tid];
  float o[4] = {g4.x * (v[0] - mean) * inv + t4.x,
                g4.y * (v[1] - mean) * inv + t4.y,
                g4.z * (v[2] - mean) * inv + t4.z,
                g4.w * (v[3] - mean) * inv + t4.w};
  ((float4*)(outF + row * D))[tid] = *(const float4*)o;
  if (WF16) {
    _Float16 h[4] = {(_Float16)o[0], (_Float16)o[1],
                     (_Float16)o[2], (_Float16)o[3]};
    ((uint2*)(outH + row * D))[tid] = *(const uint2*)h;
  }
}

// ---------------------------------------------------------------------------
// kernel_launch
// ---------------------------------------------------------------------------
extern "C" void kernel_launch(void* const* d_in, const int* in_sizes, int n_in,
                              void* d_out, int out_size, void* d_ws,
                              size_t ws_size, hipStream_t stream) {
  (void)in_sizes; (void)n_in; (void)out_size; (void)ws_size;
  constexpr int B = 32, S = 512, D = 1024, H = 16, F = 4096;
  const size_t BS = (size_t)B * S;

  const float* x    = (const float*)d_in[0];
  const int*   mask = (const int*)d_in[1];
  const float* adj  = (const float*)d_in[2];
  // d_in[3] = num_head (shapes fixed by setup_inputs)
  const float* Wq = (const float*)d_in[4];  const float* bq = (const float*)d_in[5];
  const float* Wk = (const float*)d_in[6];  const float* bk = (const float*)d_in[7];
  const float* Wv = (const float*)d_in[8];  const float* bv = (const float*)d_in[9];
  const float* Wo = (const float*)d_in[10]; const float* bo = (const float*)d_in[11];
  const float* W1 = (const float*)d_in[12]; const float* b1 = (const float*)d_in[13];
  const float* W2 = (const float*)d_in[14]; const float* b2 = (const float*)d_in[15];
  const float* gamma = (const float*)d_in[16];
  const float* beta  = (const float*)d_in[17];
  float* out = (float*)d_out;

  char* ws = (char*)d_ws;
  auto alloc = [&](size_t bytes) -> char* {
    char* p = ws; ws += (bytes + 255) & ~(size_t)255; return p;
  };
  _Float16* xh   = (_Float16*)alloc(BS * D * 2);
  _Float16* wqh  = (_Float16*)alloc((size_t)D * D * 2);
  _Float16* wkh  = (_Float16*)alloc((size_t)D * D * 2);
  _Float16* wvh  = (_Float16*)alloc((size_t)D * D * 2);
  _Float16* woh  = (_Float16*)alloc((size_t)D * D * 2);
  _Float16* w1h  = (_Float16*)alloc((size_t)D * F * 2);
  _Float16* w2h  = (_Float16*)alloc((size_t)F * D * 2);
  _Float16* q16  = (_Float16*)alloc(BS * D * 2);
  _Float16* k16  = (_Float16*)alloc(BS * D * 2);
  _Float16* v16  = (_Float16*)alloc(BS * D * 2);
  _Float16* ctxh = (_Float16*)alloc(BS * D * 2);
  float*    attn_out = (float*)alloc(BS * D * 4);
  float*    out1     = (float*)alloc(BS * D * 4);
  _Float16* out1h    = (_Float16*)alloc(BS * D * 2);
  _Float16* h1h      = (_Float16*)alloc(BS * F * 2);
  float*    ffn      = (float*)alloc(BS * D * 4);

  // 1) precision conversion (float4 granularity)
  cvt_f32_to_f16<<<2048, 256, 0, stream>>>(x,  xh,  BS * D / 4);
  cvt_f32_to_f16<<<512,  256, 0, stream>>>(Wq, wqh, (size_t)D * D / 4);
  cvt_f32_to_f16<<<512,  256, 0, stream>>>(Wk, wkh, (size_t)D * D / 4);
  cvt_f32_to_f16<<<512,  256, 0, stream>>>(Wv, wvh, (size_t)D * D / 4);
  cvt_f32_to_f16<<<512,  256, 0, stream>>>(Wo, woh, (size_t)D * D / 4);
  cvt_f32_to_f16<<<1024, 256, 0, stream>>>(W1, w1h, (size_t)D * F / 4);
  cvt_f32_to_f16<<<1024, 256, 0, stream>>>(W2, w2h, (size_t)F * D / 4);

  const dim3 blk(256);
  const dim3 gDxD(D / 64, BS / 128);  // (16, 128)
  const dim3 gDxF(F / 64, BS / 128);  // (64, 128)

  // 2) Q/K/V projections (f16 outputs feed attention WMMA)
  gemm_f16_wmma<false, true, false><<<gDxD, blk, 0, stream>>>(
      xh, wqh, bq, nullptr, q16, (int)BS, D, D);
  gemm_f16_wmma<false, true, false><<<gDxD, blk, 0, stream>>>(
      xh, wkh, bk, nullptr, k16, (int)BS, D, D);
  gemm_f16_wmma<false, true, false><<<gDxD, blk, 0, stream>>>(
      xh, wvh, bv, nullptr, v16, (int)BS, D, D);

  // 3) attention (scores + softmax + ctx), heads via strided access
  attention_wmma<<<dim3(S / 64, H, B), blk, 0, stream>>>(q16, k16, v16, adj,
                                                         mask, ctxh);

  // 4) output projection (f32 for residual+LN)
  gemm_f16_wmma<true, false, false><<<gDxD, blk, 0, stream>>>(
      ctxh, woh, bo, attn_out, nullptr, (int)BS, D, D);

  // 5) out1 = LN(x + attn_out) : f32 + f16 copies
  add_layernorm<true><<<(int)BS, blk, 0, stream>>>(x, attn_out, gamma, beta,
                                                   out1, out1h);

  // 6) FFN: gelu_erf(out1 @ W1 + b1) @ W2 + b2
  gemm_f16_wmma<false, true, true><<<gDxF, blk, 0, stream>>>(
      out1h, w1h, b1, nullptr, h1h, (int)BS, F, D);
  gemm_f16_wmma<true, false, false><<<gDxD, blk, 0, stream>>>(
      h1h, w2h, b2, ffn, nullptr, (int)BS, D, F);

  // 7) final LN(out1 + ffn) -> d_out
  add_layernorm<false><<<(int)BS, blk, 0, stream>>>(out1, ffn, gamma, beta,
                                                    out, nullptr);
}
